// EncodeProcessDecode_80487687127421
// MI455X (gfx1250) — matrix operand order repaired and checked by reading, not compiled, not run on hardware
//
#include <hip/hip_runtime.h>
#include <math.h>

typedef __bf16 bf16;
typedef __attribute__((ext_vector_type(16))) __bf16 v16bf;
typedef __attribute__((ext_vector_type(8)))  __bf16 v8bf;
typedef __attribute__((ext_vector_type(8)))  float  v8f;

#define LN_EPS 1e-5f
#define H 128
#define TW 5
#define PSTEPS 3

// ---------------------------------------------------------------------------
// WMMA helpers (CDNA5 wave32, v_wmma_f32_16x16x32_bf16)
// ---------------------------------------------------------------------------
__device__ __forceinline__ v8f wmma_bf16(v16bf a, v16bf b, v8f c) {
  return __builtin_amdgcn_wmma_f32_16x16x32_bf16(
      /*neg_a=*/false, a, /*neg_b=*/false, b,
      /*c_mod=*/(short)0, c, /*reuse_a=*/false, /*reuse_b=*/false);
}

// A-matrix 16x32 bf16 fragment from LDS (row-major, `stride` bf16 elems).
// ISA 7.12.2: lane L holds row (L&15); half = (L>>4):
//   elems 0..7  = K[kofs + half*8 .. +7], elems 8..15 = K[kofs+16+half*8 .. +7]
__device__ __forceinline__ v16bf load_a_lds(const bf16* base, int stride,
                                            int kofs, int lane) {
  const int row  = lane & 15;
  const int half = lane >> 4;
  const bf16* p = base + row * stride + kofs + half * 8;
  v8bf lo = *(const v8bf*)(p);
  v8bf hi = *(const v8bf*)(p + 16);
  v16bf a;
#pragma unroll
  for (int i = 0; i < 8; ++i) { a[i] = lo[i]; a[i + 8] = hi[i]; }
  return a;
}

// B-matrix 32x16 bf16 fragment from global. Weights stored transposed WT[n][k]
// (row length K); each lane loads K[kofs+half*8..+7] and K[kofs+16+half*8..+7].
__device__ __forceinline__ v16bf load_b_w(const bf16* WT, int K, int colbase,
                                          int kofs, int lane) {
  const int col   = colbase + (lane & 15);
  const int khalf = lane >> 4;
  const bf16* p = WT + (size_t)col * K + kofs + khalf * 8;
  v8bf lo = *(const v8bf*)(p);
  v8bf hi = *(const v8bf*)(p + 16);
  v16bf b;
#pragma unroll
  for (int i = 0; i < 8; ++i) { b[i] = lo[i]; b[i + 8] = hi[i]; }
  return b;
}

__device__ __forceinline__ void atomic_add_f(float* p, float v) {
  __hip_atomic_fetch_add(p, v, __ATOMIC_RELAXED, __HIP_MEMORY_SCOPE_AGENT);
}

// Async direct global->LDS 16B copy (gfx1250 GLOBAL_LOAD_ASYNC_TO_LDS_B128,
// tracked by ASYNCcnt). LDS operand = low 32 bits of generic shared pointer.
__device__ __forceinline__ void async_b128(void* lds, const void* g) {
  unsigned l = (unsigned)(size_t)lds;
  asm volatile("global_load_async_to_lds_b128 %0, %1, off"
               :: "v"(l), "v"(g) : "memory");
}
__device__ __forceinline__ void wait_async0() {
  asm volatile("s_wait_asynccnt 0x0" ::: "memory");
}

// LayerNorm stats for one 128-wide row owned by a 16-thread group (each thread
// holds 8 elements). xor-shuffle masks 1..8 stay inside the 16-lane half-wave.
__device__ __forceinline__ void ln_stats(const float x[8], float& m, float& inv) {
  float s = 0.f, s2 = 0.f;
#pragma unroll
  for (int j = 0; j < 8; ++j) { s += x[j]; s2 += x[j] * x[j]; }
#pragma unroll
  for (int t = 1; t < 16; t <<= 1) {
    s  += __shfl_xor(s, t, 32);
    s2 += __shfl_xor(s2, t, 32);
  }
  m = s * (1.f / H);
  float var = s2 * (1.f / H) - m * m;
  inv = rsqrtf(fmaxf(var, 0.f) + LN_EPS);
}

// ---------------------------------------------------------------------------
// Weight prep: fp32 [Kreal][Ncols] -> bf16 transposed [Ncols][Kpad] (K zero-pad)
// ---------------------------------------------------------------------------
__global__ void transpose_w_kernel(const float* __restrict__ src,
                                   bf16* __restrict__ dst,
                                   int Kreal, int Kpad, int Ncols) {
  int t = blockIdx.x * blockDim.x + threadIdx.x;
  if (t >= Kpad * Ncols) return;
  int n = t / Kpad, k = t % Kpad;
  dst[t] = (k < Kreal) ? (bf16)src[(size_t)k * Ncols + n] : (bf16)0.f;
}

// ---------------------------------------------------------------------------
// Feature builders (normalized, bf16, padded to 32 for one K=32 WMMA chunk)
// ---------------------------------------------------------------------------
__global__ void node_feat_kernel(int N, const float* __restrict__ u,
                                 const float* __restrict__ up,
                                 const float* __restrict__ q,
                                 const float* __restrict__ qp,
                                 const float* __restrict__ qn,
                                 const float* __restrict__ mean,
                                 const float* __restrict__ stdv,
                                 bf16* __restrict__ nf) {
  int n = blockIdx.x * blockDim.x + threadIdx.x;
  if (n >= N) return;
  float f[9];
  f[0] = u[n]; f[1] = u[n] - up[n];
  f[2] = q[n]; f[3] = q[n] - qp[n];
#pragma unroll
  for (int t = 0; t < TW; ++t) f[4 + t] = qn[(size_t)n * TW + t];
  bf16* dst = nf + (size_t)n * 32;
#pragma unroll
  for (int j = 0; j < 9; ++j) dst[j] = (bf16)((f[j] - mean[j]) / stdv[j]);
#pragma unroll
  for (int j = 9; j < 32; ++j) dst[j] = (bf16)0.f;
}

__global__ void edge_feat_kernel(int E, const int* __restrict__ ei,
                                 const float* __restrict__ pos,
                                 const float* __restrict__ u,
                                 const float* __restrict__ mean,
                                 const float* __restrict__ stdv,
                                 bf16* __restrict__ ef) {
  int e = blockIdx.x * blockDim.x + threadIdx.x;
  if (e >= E) return;
  int s = ei[e], r = ei[E + e];
  float rx = pos[2 * s] - pos[2 * r];
  float ry = pos[2 * s + 1] - pos[2 * r + 1];
  float dist = sqrtf(rx * rx + ry * ry);
  float relt = u[s] - u[r];
  float f[4] = {rx, ry, dist, relt};
  bf16* dst = ef + (size_t)e * 32;
#pragma unroll
  for (int j = 0; j < 4; ++j) dst[j] = (bf16)((f[j] - mean[j]) / stdv[j]);
#pragma unroll
  for (int j = 4; j < 32; ++j) dst[j] = (bf16)0.f;
}

// ---------------------------------------------------------------------------
// Encoder MLP: Y = LN(ReLU(X@W1+b1)@W2+b2), X is [M][32] bf16 (K-padded).
// Block = 256 threads = 8 waves; 16-row tile, each wave owns 16 output cols.
// ---------------------------------------------------------------------------
__global__ void __launch_bounds__(256)
encode_mlp_kernel(const bf16* __restrict__ Xb, int M,
                  const bf16* __restrict__ W1T, const float* __restrict__ b1,
                  const bf16* __restrict__ W2T, const float* __restrict__ b2,
                  const float* __restrict__ g, const float* __restrict__ bb,
                  float* __restrict__ Yf, bf16* __restrict__ Yb) {
  __shared__ bf16 sA[16 * 32];
  __shared__ bf16 sH[16 * H];
  __shared__ float sO[16 * H];
  const int tid = threadIdx.x;
  const int lane = tid & 31;
  const int wave = tid >> 5;
  const int row0 = blockIdx.x * 16;

  if (tid < 64) {  // async gather X tile (16 rows x 64B)
    int r = tid >> 2, cb = (tid & 3) * 8;
    int gr = row0 + r;
    if (gr < M) {
      async_b128(sA + r * 32 + cb, Xb + (size_t)gr * 32 + cb);
    } else {
      v8bf z = {};
      *(v8bf*)(sA + r * 32 + cb) = z;
    }
  }
  wait_async0();
  __syncthreads();

  const int ct  = wave * 16;
  const int col = ct + (lane & 15);
  const int rb  = (lane >> 4) * 8;
  {  // layer 1 (K=32, one WMMA) + ReLU -> bf16 LDS
    v8f acc = {};
    v16bf a = load_a_lds(sA, 32, 0, lane);
    v16bf b = load_b_w(W1T, 32, ct, 0, lane);
    acc = wmma_bf16(a, b, acc);
    float bias = b1[col];
    bf16* hp = sH + rb * H + col;
#pragma unroll
    for (int v = 0; v < 8; ++v) hp[v * H] = (bf16)fmaxf(acc[v] + bias, 0.f);
  }
  __syncthreads();
  {  // layer 2 (K=128, 4 WMMAs)
    v8f acc = {};
#pragma unroll
    for (int kc = 0; kc < 4; ++kc) {
      v16bf a = load_a_lds(sH, H, kc * 32, lane);
      v16bf b = load_b_w(W2T, H, ct, kc * 32, lane);
      acc = wmma_bf16(a, b, acc);
    }
    float bias = b2[col];
    float* op = sO + rb * H + col;
#pragma unroll
    for (int v = 0; v < 8; ++v) op[v * H] = acc[v] + bias;
  }
  __syncthreads();
  {  // LN + output (shuffle stats over the row's 16-thread group)
    int r = tid >> 4;
    int cb = (tid & 15) * 8;
    int gr = row0 + r;
    float x[8];
    const float* orow = sO + r * H + cb;
#pragma unroll
    for (int j = 0; j < 8; ++j) x[j] = orow[j];
    float m, inv;
    ln_stats(x, m, inv);
    if (gr < M) {
      float o[8];
      v8bf ob;
#pragma unroll
      for (int j = 0; j < 8; ++j) {
        int c = cb + j;
        o[j] = (x[j] - m) * inv * g[c] + bb[c];
        ob[j] = (bf16)o[j];
      }
      *(float4*)(Yf + (size_t)gr * H + cb)     = *(float4*)(o);
      *(float4*)(Yf + (size_t)gr * H + cb + 4) = *(float4*)(o + 4);
      *(v8bf*)(Yb + (size_t)gr * H + cb) = ob;
    }
  }
}

// ---------------------------------------------------------------------------
// Fused edge processor: per 16-edge tile,
//   msg   = LN(MLP(cat[x_r, x_s, e]))      -> atomic scatter-add at receiver
//   new_e = LN(MLP(cat[x_s, x_r, e])) + e  -> in-place edge state update
// B tiles shared between the two variants: 2 WMMAs per weight load.
// ---------------------------------------------------------------------------
__global__ void __launch_bounds__(256)
edge_proc_kernel(const int* __restrict__ ei, int E,
                 const bf16* __restrict__ xb,
                 float* __restrict__ eh_f, bf16* __restrict__ eh_b,
                 const bf16* __restrict__ W1T, const float* __restrict__ b1,
                 const bf16* __restrict__ W2T, const float* __restrict__ b2,
                 const float* __restrict__ g, const float* __restrict__ bb,
                 float* __restrict__ aggr) {
  __shared__ bf16 sXR[16 * H], sXS[16 * H], sE[16 * H];
  __shared__ bf16 sHm[16 * H], sHu[16 * H];
  __shared__ float sOm[16 * H], sOu[16 * H];
  __shared__ int sIdx[32];
  const int tid = threadIdx.x;
  const int lane = tid & 31;
  const int wave = tid >> 5;
  const int row0 = blockIdx.x * 16;

  if (tid < 16) {
    int e = row0 + tid;
    sIdx[tid]      = (e < E) ? ei[e]     : 0;  // sender j
    sIdx[16 + tid] = (e < E) ? ei[E + e] : 0;  // receiver i
  }
  __syncthreads();
  {  // async gather x_s, x_r, e tiles (16B per thread per tile)
    int r = tid >> 4;
    int cb = (tid & 15) * 8;
    int e = row0 + r;
    int s = sIdx[r], rc = sIdx[16 + r];
    async_b128(sXS + r * H + cb, xb + (size_t)s  * H + cb);
    async_b128(sXR + r * H + cb, xb + (size_t)rc * H + cb);
    if (e < E) {
      async_b128(sE + r * H + cb, eh_b + (size_t)e * H + cb);
    } else {
      v8bf z = {};
      *(v8bf*)(sE + r * H + cb) = z;
    }
  }
  wait_async0();
  __syncthreads();

  const int ct  = wave * 16;
  const int col = ct + (lane & 15);
  const int rb  = (lane >> 4) * 8;
  {  // layer 1: K=384 = [block0|block1|block2] x 4 chunks each
    v8f accm = {}, accu = {};
#pragma unroll
    for (int kc = 0; kc < 12; ++kc) {
      int seg = kc >> 2;
      int kloc = (kc & 3) * 32;
      const bf16* am = (seg == 0) ? sXR : (seg == 1) ? sXS : sE;  // msg order
      const bf16* au = (seg == 0) ? sXS : (seg == 1) ? sXR : sE;  // update order
      v16bf b = load_b_w(W1T, 384, ct, kc * 32, lane);
      v16bf a0 = load_a_lds(am, H, kloc, lane);
      accm = wmma_bf16(a0, b, accm);
      v16bf a1 = load_a_lds(au, H, kloc, lane);
      accu = wmma_bf16(a1, b, accu);
    }
    float bias = b1[col];
    bf16* hm = sHm + rb * H + col;
    bf16* hu = sHu + rb * H + col;
#pragma unroll
    for (int v = 0; v < 8; ++v) {
      hm[v * H] = (bf16)fmaxf(accm[v] + bias, 0.f);
      hu[v * H] = (bf16)fmaxf(accu[v] + bias, 0.f);
    }
  }
  __syncthreads();
  {  // layer 2: K=128, shared B tiles again
    v8f accm = {}, accu = {};
#pragma unroll
    for (int kc = 0; kc < 4; ++kc) {
      v16bf b = load_b_w(W2T, H, ct, kc * 32, lane);
      v16bf a0 = load_a_lds(sHm, H, kc * 32, lane);
      accm = wmma_bf16(a0, b, accm);
      v16bf a1 = load_a_lds(sHu, H, kc * 32, lane);
      accu = wmma_bf16(a1, b, accu);
    }
    float bias = b2[col];
    float* om = sOm + rb * H + col;
    float* ou = sOu + rb * H + col;
#pragma unroll
    for (int v = 0; v < 8; ++v) {
      om[v * H] = accm[v] + bias;
      ou[v * H] = accu[v] + bias;
    }
  }
  __syncthreads();
  {  // LN (shuffle stats) + scatter messages + in-place edge residual update
    int r = tid >> 4;
    int cb = (tid & 15) * 8;
    int e = row0 + r;
    float xm[8], xu[8];
    const float* pm = sOm + r * H + cb;
    const float* pu = sOu + r * H + cb;
#pragma unroll
    for (int j = 0; j < 8; ++j) { xm[j] = pm[j]; xu[j] = pu[j]; }
    float mM, iM, mU, iU;
    ln_stats(xm, mM, iM);
    ln_stats(xu, mU, iU);
    if (e < E) {
      int rc = sIdx[16 + r];
      float ef0[8];
      *(float4*)(ef0)     = *(const float4*)(eh_f + (size_t)e * H + cb);
      *(float4*)(ef0 + 4) = *(const float4*)(eh_f + (size_t)e * H + cb + 4);
      float upd[8];
      v8bf ub;
#pragma unroll
      for (int j = 0; j < 8; ++j) {
        int c = cb + j;
        float msg = (xm[j] - mM) * iM * g[c] + bb[c];
        atomic_add_f(&aggr[(size_t)rc * H + c], msg);
        upd[j] = (xu[j] - mU) * iU * g[c] + bb[c] + ef0[j];
        ub[j] = (bf16)upd[j];
      }
      *(float4*)(eh_f + (size_t)e * H + cb)     = *(float4*)(upd);
      *(float4*)(eh_f + (size_t)e * H + cb + 4) = *(float4*)(upd + 4);
      *(v8bf*)(eh_b + (size_t)e * H + cb) = ub;
    }
  }
}

// ---------------------------------------------------------------------------
// Node processor: new_x = LN(MLP(cat[aggr, x])) + x, in place.
// ---------------------------------------------------------------------------
__global__ void __launch_bounds__(256)
node_proc_kernel(int N, const float* __restrict__ aggr,
                 float* __restrict__ xh_f, bf16* __restrict__ xh_b,
                 const bf16* __restrict__ W1T, const float* __restrict__ b1,
                 const bf16* __restrict__ W2T, const float* __restrict__ b2,
                 const float* __restrict__ g, const float* __restrict__ bb) {
  __shared__ bf16 sA[16 * 256];
  __shared__ bf16 sH[16 * H];
  __shared__ float sO[16 * H];
  const int tid = threadIdx.x;
  const int lane = tid & 31;
  const int wave = tid >> 5;
  const int row0 = blockIdx.x * 16;
  {  // gather [aggr (fp32->bf16) | x (async)] tile
    int r = tid >> 4;
    int cb = (tid & 15) * 8;
    int n = row0 + r;
    if (n < N) {
      float av[8];
      *(float4*)(av)     = *(const float4*)(aggr + (size_t)n * H + cb);
      *(float4*)(av + 4) = *(const float4*)(aggr + (size_t)n * H + cb + 4);
      v8bf bv;
#pragma unroll
      for (int j = 0; j < 8; ++j) bv[j] = (bf16)av[j];
      *(v8bf*)(sA + r * 256 + cb) = bv;
      async_b128(sA + r * 256 + 128 + cb, xh_b + (size_t)n * H + cb);
    } else {
      v8bf z = {};
      *(v8bf*)(sA + r * 256 + cb) = z;
      *(v8bf*)(sA + r * 256 + 128 + cb) = z;
    }
  }
  wait_async0();
  __syncthreads();
  const int ct  = wave * 16;
  const int col = ct + (lane & 15);
  const int rb  = (lane >> 4) * 8;
  {  // layer 1: K=256
    v8f acc = {};
#pragma unroll
    for (int kc = 0; kc < 8; ++kc) {
      v16bf a = load_a_lds(sA, 256, kc * 32, lane);
      v16bf b = load_b_w(W1T, 256, ct, kc * 32, lane);
      acc = wmma_bf16(a, b, acc);
    }
    float bias = b1[col];
    bf16* hp = sH + rb * H + col;
#pragma unroll
    for (int v = 0; v < 8; ++v) hp[v * H] = (bf16)fmaxf(acc[v] + bias, 0.f);
  }
  __syncthreads();
  {  // layer 2: K=128
    v8f acc = {};
#pragma unroll
    for (int kc = 0; kc < 4; ++kc) {
      v16bf a = load_a_lds(sH, H, kc * 32, lane);
      v16bf b = load_b_w(W2T, H, ct, kc * 32, lane);
      acc = wmma_bf16(a, b, acc);
    }
    float bias = b2[col];
    float* op = sO + rb * H + col;
#pragma unroll
    for (int v = 0; v < 8; ++v) op[v * H] = acc[v] + bias;
  }
  __syncthreads();
  {  // LN + residual + output
    int r = tid >> 4;
    int cb = (tid & 15) * 8;
    int n = row0 + r;
    float x[8];
    const float* orow = sO + r * H + cb;
#pragma unroll
    for (int j = 0; j < 8; ++j) x[j] = orow[j];
    float m, inv;
    ln_stats(x, m, inv);
    if (n < N) {
      float res[8];
      *(float4*)(res)     = *(const float4*)(xh_f + (size_t)n * H + cb);
      *(float4*)(res + 4) = *(const float4*)(xh_f + (size_t)n * H + cb + 4);
      float o[8];
      v8bf ob;
#pragma unroll
      for (int j = 0; j < 8; ++j) {
        int c = cb + j;
        o[j] = (x[j] - m) * inv * g[c] + bb[c] + res[j];
        ob[j] = (bf16)o[j];
      }
      *(float4*)(xh_f + (size_t)n * H + cb)     = *(float4*)(o);
      *(float4*)(xh_f + (size_t)n * H + cb + 4) = *(float4*)(o + 4);
      *(v8bf*)(xh_b + (size_t)n * H + cb) = ob;
    }
  }
}

// ---------------------------------------------------------------------------
// Decoder: h = swish(x@W1+b1) (WMMA, 128->64), delta = h@W2+b2, out = dt*delta
// ---------------------------------------------------------------------------
__global__ void __launch_bounds__(256)
decode_kernel(int N, const bf16* __restrict__ xb,
              const bf16* __restrict__ W1T, const float* __restrict__ b1,
              const float* __restrict__ w2, const float* __restrict__ b2,
              float* __restrict__ out) {
  __shared__ bf16 sA[16 * H];
  __shared__ float sHd[16 * 64];
  const int tid = threadIdx.x;
  const int lane = tid & 31;
  const int wave = tid >> 5;
  const int row0 = blockIdx.x * 16;
  {
    int r = tid >> 4;
    int cb = (tid & 15) * 8;
    int n = row0 + r;
    if (n < N) {
      async_b128(sA + r * H + cb, xb + (size_t)n * H + cb);
    } else {
      v8bf z = {};
      *(v8bf*)(sA + r * H + cb) = z;
    }
  }
  wait_async0();
  __syncthreads();
  if (wave < 4) {  // 64 output cols = 4 waves
    const int ct  = wave * 16;
    const int col = ct + (lane & 15);
    const int rb  = (lane >> 4) * 8;
    v8f acc = {};
#pragma unroll
    for (int kc = 0; kc < 4; ++kc) {
      v16bf a = load_a_lds(sA, H, kc * 32, lane);
      v16bf b = load_b_w(W1T, H, ct, kc * 32, lane);
      acc = wmma_bf16(a, b, acc);
    }
    float bias = b1[col];
    float* hp = sHd + rb * 64 + col;
#pragma unroll
    for (int v = 0; v < 8; ++v) {
      float h = acc[v] + bias;
      hp[v * 64] = h / (1.f + __expf(-h));  // swish
    }
  }
  __syncthreads();
  if (tid < 16 * TW) {
    int r = tid / TW, o = tid % TW;
    int n = row0 + r;
    if (n < N) {
      float acc = b2[o];
      for (int k = 0; k < 64; ++k) acc += sHd[r * 64 + k] * w2[k * TW + o];
      out[(size_t)n * TW + o] = (float)(o + 1) * acc;
    }
  }
}

// ---------------------------------------------------------------------------
// Launch
// ---------------------------------------------------------------------------
extern "C" void kernel_launch(void* const* d_in, const int* in_sizes, int n_in,
                              void* d_out, int out_size, void* d_ws, size_t ws_size,
                              hipStream_t stream) {
  (void)n_in; (void)out_size; (void)ws_size;
  const float* temperature      = (const float*)d_in[0];
  const float* temperature_prev = (const float*)d_in[1];
  const float* heat_source      = (const float*)d_in[2];
  const float* heat_source_prev = (const float*)d_in[3];
  const float* heat_source_next = (const float*)d_in[4];
  const float* mesh_pos         = (const float*)d_in[5];
  const int*   edge_index       = (const int*)d_in[6];
  const float* node_mean = (const float*)d_in[7];
  const float* node_std  = (const float*)d_in[8];
  const float* edge_mean = (const float*)d_in[9];
  const float* edge_std  = (const float*)d_in[10];
  const float* ne_w1 = (const float*)d_in[11]; const float* ne_b1 = (const float*)d_in[12];
  const float* ne_w2 = (const float*)d_in[13]; const float* ne_b2 = (const float*)d_in[14];
  const float* ne_g  = (const float*)d_in[15]; const float* ne_be = (const float*)d_in[16];
  const float* ee_w1 = (const float*)d_in[17]; const float* ee_b1 = (const float*)d_in[18];
  const float* ee_w2 = (const float*)d_in[19]; const float* ee_b2 = (const float*)d_in[20];
  const float* ee_g  = (const float*)d_in[21]; const float* ee_be = (const float*)d_in[22];
  const float* pe_w1 = (const float*)d_in[23]; const float* pe_b1 = (const float*)d_in[24];
  const float* pe_w2 = (const float*)d_in[25]; const float* pe_b2 = (const float*)d_in[26];
  const float* pe_g  = (const float*)d_in[27]; const float* pe_be = (const float*)d_in[28];
  const float* pn_w1 = (const float*)d_in[29]; const float* pn_b1 = (const float*)d_in[30];
  const float* pn_w2 = (const float*)d_in[31]; const float* pn_b2 = (const float*)d_in[32];
  const float* pn_g  = (const float*)d_in[33]; const float* pn_be = (const float*)d_in[34];
  const float* dec_w1 = (const float*)d_in[35]; const float* dec_b1 = (const float*)d_in[36];
  const float* dec_w2 = (const float*)d_in[37]; const float* dec_b2 = (const float*)d_in[38];

  const int N = in_sizes[0];
  const int E = in_sizes[6] / 2;

  char* cur = (char*)d_ws;
  auto carve = [&](size_t bytes) -> char* {
    char* p = cur;
    cur += (bytes + 255) & ~(size_t)255;
    return p;
  };
  bf16* neW1T = (bf16*)carve(128 * 32 * sizeof(bf16));
  bf16* neW2T = (bf16*)carve(128 * 128 * sizeof(bf16));
  bf16* eeW1T = (bf16*)carve(128 * 32 * sizeof(bf16));
  bf16* eeW2T = (bf16*)carve(128 * 128 * sizeof(bf16));
  bf16* peW1T = (bf16*)carve((size_t)PSTEPS * 128 * 384 * sizeof(bf16));
  bf16* peW2T = (bf16*)carve((size_t)PSTEPS * 128 * 128 * sizeof(bf16));
  bf16* pnW1T = (bf16*)carve((size_t)PSTEPS * 128 * 256 * sizeof(bf16));
  bf16* pnW2T = (bf16*)carve((size_t)PSTEPS * 128 * 128 * sizeof(bf16));
  bf16* decW1T = (bf16*)carve(64 * 128 * sizeof(bf16));
  bf16* nf    = (bf16*)carve((size_t)N * 32 * sizeof(bf16));
  bf16* ef    = (bf16*)carve((size_t)E * 32 * sizeof(bf16));
  float* xh_f = (float*)carve((size_t)N * H * sizeof(float));
  bf16*  xh_b = (bf16*)carve((size_t)N * H * sizeof(bf16));
  float* eh_f = (float*)carve((size_t)E * H * sizeof(float));
  bf16*  eh_b = (bf16*)carve((size_t)E * H * sizeof(bf16));
  float* aggr = (float*)carve((size_t)N * H * sizeof(float));

  auto tlaunch = [&](const float* src, bf16* dst, int Kreal, int Kpad, int Ncols) {
    int total = Kpad * Ncols;
    transpose_w_kernel<<<(total + 255) / 256, 256, 0, stream>>>(src, dst, Kreal, Kpad, Ncols);
  };
  tlaunch(ne_w1, neW1T, 9, 32, 128);
  tlaunch(ne_w2, neW2T, 128, 128, 128);
  tlaunch(ee_w1, eeW1T, 4, 32, 128);
  tlaunch(ee_w2, eeW2T, 128, 128, 128);
  for (int i = 0; i < PSTEPS; ++i) {
    tlaunch(pe_w1 + (size_t)i * 384 * 128, peW1T + (size_t)i * 128 * 384, 384, 384, 128);
    tlaunch(pe_w2 + (size_t)i * 128 * 128, peW2T + (size_t)i * 128 * 128, 128, 128, 128);
    tlaunch(pn_w1 + (size_t)i * 256 * 128, pnW1T + (size_t)i * 128 * 256, 256, 256, 128);
    tlaunch(pn_w2 + (size_t)i * 128 * 128, pnW2T + (size_t)i * 128 * 128, 128, 128, 128);
  }
  tlaunch(dec_w1, decW1T, 128, 128, 64);

  node_feat_kernel<<<(N + 255) / 256, 256, 0, stream>>>(
      N, temperature, temperature_prev, heat_source, heat_source_prev,
      heat_source_next, node_mean, node_std, nf);
  edge_feat_kernel<<<(E + 255) / 256, 256, 0, stream>>>(
      E, edge_index, mesh_pos, temperature, edge_mean, edge_std, ef);

  encode_mlp_kernel<<<(N + 15) / 16, 256, 0, stream>>>(
      nf, N, neW1T, ne_b1, neW2T, ne_b2, ne_g, ne_be, xh_f, xh_b);
  encode_mlp_kernel<<<(E + 15) / 16, 256, 0, stream>>>(
      ef, E, eeW1T, ee_b1, eeW2T, ee_b2, ee_g, ee_be, eh_f, eh_b);

  for (int i = 0; i < PSTEPS; ++i) {
    hipMemsetAsync(aggr, 0, (size_t)N * H * sizeof(float), stream);
    edge_proc_kernel<<<(E + 15) / 16, 256, 0, stream>>>(
        edge_index, E, xh_b, eh_f, eh_b,
        peW1T + (size_t)i * 128 * 384, pe_b1 + (size_t)i * H,
        peW2T + (size_t)i * 128 * 128, pe_b2 + (size_t)i * H,
        pe_g + (size_t)i * H, pe_be + (size_t)i * H, aggr);
    node_proc_kernel<<<(N + 15) / 16, 256, 0, stream>>>(
        N, aggr, xh_f, xh_b,
        pnW1T + (size_t)i * 128 * 256, pn_b1 + (size_t)i * H,
        pnW2T + (size_t)i * 128 * 128, pn_b2 + (size_t)i * H,
        pn_g + (size_t)i * H, pn_be + (size_t)i * H);
  }
  decode_kernel<<<(N + 15) / 16, 256, 0, stream>>>(
      N, xh_b, decW1T, dec_b1, dec_w2, dec_b2, (float*)d_out);
}
